// DiscreteDiffusionActionHead_68719477558
// MI455X (gfx1250) — compile-verified, matrix operand order checked by reading, not run on hardware
//
#include <hip/hip_runtime.h>
#include <hip/hip_bf16.h>

typedef __attribute__((ext_vector_type(16))) __bf16 v16bf;
typedef __attribute__((ext_vector_type(8)))  __bf16 bf16x8;
typedef __attribute__((ext_vector_type(8)))  float  v8f;

#define LAYERS 24
#define DIM    896
#define NHD    8
#define HD     112
#define HDP    128
#define TQ     56
#define NVIS   512
#define NADP   64
#define NTOK   576
#define VOCAB  256
#define BB     8
#define KEYS   633
#define KEYP   640
#define ROWS_X (BB*TQ)       // 448
#define ROWS_A (BB*(NADP+1)) // 520
#define ROWS_T (BB*NVIS)     // 4096
#define ZBH    (BB*NHD)      // 64

// ---------------------------------------------------------------------------
// Block reduction helper (256 threads)
// ---------------------------------------------------------------------------
__device__ __forceinline__ float block_reduce(float v, float* red, int is_max) {
    int tid = threadIdx.x;
    red[tid] = v;
    __syncthreads();
    for (int s = 128; s > 0; s >>= 1) {
        if (tid < s) {
            float a = red[tid], b = red[tid + s];
            red[tid] = is_max ? fmaxf(a, b) : (a + b);
        }
        __syncthreads();
    }
    float r = red[0];
    __syncthreads();
    return r;
}

// Fragment-space LDS mapping for 16x16x32 bf16 WMMA:
//   A element (row r in tile, k) -> [r>>4][(r&15) + 16*((k>>3)&1)][(k&7) + 8*(k>>4)]
//   B element (k, col n in tile) -> [n>>4][(n&15) + 16*((k>>3)&1)][(k&7) + 8*(k>>4)]
// A lane's whole 16-element fragment is contiguous -> single 32B LDS load.

// ---------------------------------------------------------------------------
// Generic GEMM: C[M,N] = A[M,K](f32, batch-strided rows) @ W[K,N](f32) + bias,
// optional relu. bf16 WMMA f32-acc. Block 256 thr (8 waves), tile 64x128, BK=32.
// Requires K%32==0, N%128==0.
// ---------------------------------------------------------------------------
__global__ __launch_bounds__(256)
void gemm_f32(const float* __restrict__ A, const float* __restrict__ W,
              const float* __restrict__ bias, float* __restrict__ C,
              int M, int N, int K, int a_rpb, long long a_bstride, int relu)
{
    __shared__ alignas(32) __bf16 AsF[4][32][16];   // 4 row-tiles
    __shared__ alignas(32) __bf16 BsF[8][32][16];   // 8 col-tiles
    int tid  = threadIdx.x;
    int m0   = blockIdx.y * 64;
    int n0   = blockIdx.x * 128;
    int w    = tid >> 5, lane = tid & 31;
    int wr   = w & 3, wc = w >> 2;                  // wave: rows wr*16, cols wc*64

    // A loader: 64x32 tile, 8 f32/thread (one k-chunk of 8)
    int arow  = tid >> 2, akc = (tid & 3) << 3;
    int alane = (arow & 15) + (((akc >> 3) & 1) << 4);
    int asb   = (akc >> 4) << 3;
    // B loader: 32x128 tile, 16 f32/thread (fixed n, k-chunks [bkc,bkc+8),[bkc+8,bkc+16))
    int bn    = tid & 127, bkc = (tid >> 7) << 4;   // bkc = 0 or 16
    int btile = bn >> 4;
    int blane = bn & 15;
    int bsb   = bkc >> 1;                           // 0 or 8

    int gr = m0 + arow;
    const float* arowp = nullptr;
    if (gr < M)
        arowp = A + (size_t)(gr / a_rpb) * a_bstride + (size_t)(gr % a_rpb) * K;
    const float* bbase = W + n0 + bn;

    v8f acc[4] = {v8f{}, v8f{}, v8f{}, v8f{}};

    for (int k0 = 0; k0 < K; k0 += 32) {
        // ---- stage A ----
        bf16x8 av = {};
        if (arowp) {
            const float* ap = arowp + k0 + akc;
            #pragma unroll
            for (int j = 0; j < 8; ++j) av[j] = (__bf16)ap[j];
        }
        *(bf16x8*)&AsF[arow >> 4][alane][asb] = av;
        // ---- stage B ----
        {
            const float* bp = bbase + (size_t)(k0 + bkc) * N;
            bf16x8 b0v, b1v;
            #pragma unroll
            for (int j = 0; j < 8; ++j) b0v[j] = (__bf16)bp[(size_t)j * N];
            #pragma unroll
            for (int j = 0; j < 8; ++j) b1v[j] = (__bf16)bp[(size_t)(j + 8) * N];
            *(bf16x8*)&BsF[btile][blane][bsb]      = b0v;
            *(bf16x8*)&BsF[btile][blane + 16][bsb] = b1v;
        }
        if (k0 + 32 < K) {
            if (arowp) __builtin_prefetch((const void*)(arowp + k0 + 32 + akc), 0, 0);
            __builtin_prefetch((const void*)(bbase + (size_t)(k0 + 32 + bkc) * N), 0, 0);
        }
        __syncthreads();

        v16bf a = *(const v16bf*)AsF[wr][lane];
        #pragma unroll
        for (int t = 0; t < 4; ++t) {
            v16bf b = *(const v16bf*)BsF[wc * 4 + t][lane];
            acc[t] = __builtin_amdgcn_wmma_f32_16x16x32_bf16(
                false, a, false, b, (short)0, acc[t], false, false);
        }
        __syncthreads();
    }

    int mo = (lane >> 4) << 3;
    int nl = lane & 15;
    #pragma unroll
    for (int t = 0; t < 4; ++t) {
        int gc = n0 + wc * 64 + t * 16 + nl;
        float bia = bias[gc];
        #pragma unroll
        for (int r = 0; r < 8; ++r) {
            int grow = m0 + wr * 16 + mo + r;
            if (grow < M) {
                float v = acc[t][r] + bia;
                if (relu) v = fmaxf(v, 0.f);
                C[(size_t)grow * N + gc] = v;
            }
        }
    }
}

// ---------------------------------------------------------------------------
// Batched bf16 GEMM (grid.z batches): C[M,N](f32) = scale * A[M,K] @ B
//   btrans==0 : B stored [K,N];  btrans==1 : B stored [N,K] (C = A @ B^T)
// Block 256 thr, tile 64x64, BK=32. K%32==0, N%64==0. A row stride == K.
// ---------------------------------------------------------------------------
__global__ __launch_bounds__(256)
void gemm_bf16_batched(const __bf16* __restrict__ A, const __bf16* __restrict__ Bm,
                       float* __restrict__ C, int M, int N, int K,
                       long long aB, long long bB, long long cB,
                       int btrans, float scale)
{
    __shared__ alignas(32) __bf16 AsF[4][32][16];
    __shared__ alignas(32) __bf16 BsF[4][32][16];
    int z = blockIdx.z;
    A  += (size_t)z * aB;
    Bm += (size_t)z * bB;
    C  += (size_t)z * cB;

    int tid  = threadIdx.x;
    int m0   = blockIdx.y * 64;
    int n0   = blockIdx.x * 64;
    int w    = tid >> 5, lane = tid & 31;
    int wr   = w & 3, wc = w >> 2;                 // wave: rows wr*16, cols wc*32

    // A loader: 64x32, 8 bf16/thread (16B contiguous)
    int arow  = tid >> 2, akc = (tid & 3) << 3;
    int alane = (arow & 15) + (((akc >> 3) & 1) << 4);
    int asb   = (akc >> 4) << 3;
    // B loader: 32x64, 8 bf16/thread (fixed n, k-chunk of 8)
    int bn    = tid & 63, bkc = (tid >> 6) << 3;   // bkc in {0,8,16,24}
    int btile = bn >> 4;
    int blane = (bn & 15) + (((bkc >> 3) & 1) << 4);
    int bsb   = (bkc & 7) + ((bkc >> 4) << 3);     // = 8*(bkc>>4)

    int gr = m0 + arow;
    const __bf16* arowp = (gr < M) ? (A + (size_t)gr * K) : nullptr;

    v8f acc[2] = {v8f{}, v8f{}};

    for (int k0 = 0; k0 < K; k0 += 32) {
        bf16x8 av = {};
        if (arowp) av = *(const bf16x8*)(arowp + k0 + akc);
        *(bf16x8*)&AsF[arow >> 4][alane][asb] = av;

        bf16x8 bv;
        if (btrans) {
            bv = *(const bf16x8*)(Bm + (size_t)(n0 + bn) * K + k0 + bkc);
        } else {
            #pragma unroll
            for (int j = 0; j < 8; ++j)
                bv[j] = Bm[(size_t)(k0 + bkc + j) * N + n0 + bn];
        }
        *(bf16x8*)&BsF[btile][blane][bsb] = bv;
        __syncthreads();

        v16bf a = *(const v16bf*)AsF[wr][lane];
        #pragma unroll
        for (int t = 0; t < 2; ++t) {
            v16bf b = *(const v16bf*)BsF[wc * 2 + t][lane];
            acc[t] = __builtin_amdgcn_wmma_f32_16x16x32_bf16(
                false, a, false, b, (short)0, acc[t], false, false);
        }
        __syncthreads();
    }

    int mo = (lane >> 4) << 3;
    int nl = lane & 15;
    #pragma unroll
    for (int t = 0; t < 2; ++t) {
        int gc = n0 + wc * 32 + t * 16 + nl;
        #pragma unroll
        for (int r = 0; r < 8; ++r) {
            int grow = m0 + wr * 16 + mo + r;
            if (grow < M) C[(size_t)grow * N + gc] = acc[t][r] * scale;
        }
    }
}

// ---------------------------------------------------------------------------
// Stage projection (B,S,DIM f32) -> (B,NH,ROWS,HDP bf16) with optional
// interleaved-pair RoPE (cos index = d % 56) and optional tanh(gate) scaling.
// ---------------------------------------------------------------------------
__global__ __launch_bounds__(256)
void stage_rope(const float* __restrict__ src, __bf16* __restrict__ dst,
                int S, int ROWS, int row_off, int apply_rope,
                const float* __restrict__ gate_ptr)
{
    int r = blockIdx.x;            // b*S + s
    int b = r / S, s = r % S;
    float rg = gate_ptr ? tanhf(gate_ptr[0]) : 1.f;
    const float* xr = src + (size_t)r * DIM;
    const float LOG1E4 = 9.21034037197618f; // ln(10000)

    for (int idx = threadIdx.x; idx < NHD * HDP; idx += 256) {
        int h = idx >> 7;
        int c = idx & 127;
        float val = 0.f;
        if (c < HD) {
            float x0 = xr[h * HD + c];
            if (apply_rope) {
                int j = (c < 56) ? c : c - 56;
                float inv = __expf(-LOG1E4 * (2.f * j) / 112.f);
                float ang = (float)s * inv;
                float cs, sn;
                __sincosf(ang, &sn, &cs);
                float partner = xr[h * HD + (c ^ 1)];
                float rot = (c & 1) ? partner : -partner;
                val = x0 * cs + rot * sn;
            } else {
                val = x0;
            }
            val *= rg;
        }
        dst[(((size_t)b * NHD + h) * ROWS + row_off + s) * HDP + c] = (__bf16)val;
    }
}

// ---------------------------------------------------------------------------
// Small utility kernels
// ---------------------------------------------------------------------------
__global__ __launch_bounds__(256)
void embed_kernel(const int* __restrict__ toks, const float* __restrict__ emb,
                  float* __restrict__ x)
{
    int r = blockIdx.x;
    int tok = toks[r];
    const float* e = emb + (size_t)tok * DIM;
    for (int d = threadIdx.x; d < DIM; d += 256)
        x[(size_t)r * DIM + d] = e[d];
}

__global__ __launch_bounds__(256)
void proprio_kernel(const float* __restrict__ pr, const float* __restrict__ Wp,
                    const float* __restrict__ bp, float* __restrict__ p)
{
    int b = blockIdx.y;
    int d = blockIdx.x * 256 + threadIdx.x;
    if (d < DIM) {
        float s = bp[d];
        #pragma unroll
        for (int j = 0; j < 8; ++j) s += pr[b * 8 + j] * Wp[(size_t)j * DIM + d];
        p[(size_t)b * DIM + d] = s;
    }
}

__global__ __launch_bounds__(256)
void concat_ha_kernel(const float* __restrict__ mhs, const float* __restrict__ p,
                      float* __restrict__ ha, int layer)
{
    int r = blockIdx.x;            // 0..519
    int b = r / 65, j = r % 65;
    const float* src = (j < 64)
        ? mhs + (((size_t)b * LAYERS + layer) * NTOK + NVIS + j) * DIM
        : p + (size_t)b * DIM;
    for (int d = threadIdx.x; d < DIM; d += 256)
        ha[(size_t)r * DIM + d] = src[d];
}

__global__ __launch_bounds__(256)
void zeropad_kernel(__bf16* __restrict__ kh, __bf16* __restrict__ vh)
{
    int i = blockIdx.x * 256 + threadIdx.x;      // ZBH * 7 * HDP
    if (i >= ZBH * 7 * HDP) return;
    int z = i / (7 * HDP);
    int rest = i % (7 * HDP);
    int rr = rest / HDP, c = rest % HDP;
    size_t off = ((size_t)z * KEYP + KEYS + rr) * HDP + c;
    kh[off] = (__bf16)0.f;
    vh[off] = (__bf16)0.f;
}

__global__ __launch_bounds__(256)
void softmax_kernel(const float* __restrict__ sc, __bf16* __restrict__ at)
{
    __shared__ float red[256];
    int r = blockIdx.x;            // z*56 + q
    int z = r / TQ, q = r % TQ;
    const float* row = sc + ((size_t)z * 64 + q) * KEYP;
    __bf16* orow = at + ((size_t)z * 64 + q) * KEYP;
    float m = -1e30f;
    for (int j = threadIdx.x; j < KEYS; j += 256) m = fmaxf(m, row[j]);
    m = block_reduce(m, red, 1);
    float s = 0.f;
    for (int j = threadIdx.x; j < KEYS; j += 256) s += __expf(row[j] - m);
    s = block_reduce(s, red, 0);
    float invs = 1.f / s;
    for (int j = threadIdx.x; j < KEYP; j += 256)
        orow[j] = (__bf16)((j < KEYS) ? __expf(row[j] - m) * invs : 0.f);
}

__global__ __launch_bounds__(256)
void merge_heads_kernel(const float* __restrict__ oh, float* __restrict__ om)
{
    int r = blockIdx.x;            // 0..447
    int b = r / TQ, t = r % TQ;
    for (int d = threadIdx.x; d < DIM; d += 256) {
        int h = d / HD, c = d % HD;
        om[(size_t)r * DIM + d] = oh[(((size_t)b * NHD + h) * TQ + t) * HDP + c];
    }
}

__global__ __launch_bounds__(256)
void ln_kernel(const float* __restrict__ a, const float* __restrict__ res,
               const float* __restrict__ g, const float* __restrict__ be,
               float* __restrict__ out)
{
    __shared__ float buf[DIM];
    __shared__ float red[256];
    size_t row = blockIdx.x;
    const float* ar = a + row * DIM;
    const float* rr = res ? res + row * DIM : nullptr;
    float s = 0.f;
    for (int d = threadIdx.x; d < DIM; d += 256) {
        float v = ar[d] + (rr ? rr[d] : 0.f);
        buf[d] = v;
        s += v;
    }
    s = block_reduce(s, red, 0);
    float mean = s * (1.f / DIM);
    float sq = 0.f;
    for (int d = threadIdx.x; d < DIM; d += 256) {
        float v = buf[d] - mean;
        sq += v * v;
    }
    sq = block_reduce(sq, red, 0);
    float inv = rsqrtf(sq * (1.f / DIM) + 1e-5f);
    for (int d = threadIdx.x; d < DIM; d += 256)
        out[row * DIM + d] = (buf[d] - mean) * inv * g[d] + be[d];
}

// ---------------------------------------------------------------------------
// Host launcher
// ---------------------------------------------------------------------------
extern "C" void kernel_launch(void* const* d_in, const int* in_sizes, int n_in,
                              void* d_out, int out_size, void* d_ws, size_t ws_size,
                              hipStream_t stream)
{
    const float* mhs   = (const float*)d_in[0];
    const float* prop  = (const float*)d_in[1];
    const int*   toks  = (const int*)  d_in[2];
    const float* temb  = (const float*)d_in[3];
    const float* Wq    = (const float*)d_in[4];
    const float* bq    = (const float*)d_in[5];
    const float* Wks   = (const float*)d_in[6];
    const float* bks   = (const float*)d_in[7];
    const float* Wvs   = (const float*)d_in[8];
    const float* bvs   = (const float*)d_in[9];
    const float* Wka   = (const float*)d_in[10];
    const float* bka   = (const float*)d_in[11];
    const float* Wva   = (const float*)d_in[12];
    const float* bva   = (const float*)d_in[13];
    const float* Wkt   = (const float*)d_in[14];
    const float* bkt   = (const float*)d_in[15];
    const float* Wvt   = (const float*)d_in[16];
    const float* bvt   = (const float*)d_in[17];
    const float* Wo    = (const float*)d_in[18];
    const float* bo    = (const float*)d_in[19];
    const float* ln_g  = (const float*)d_in[20];
    const float* ln_b  = (const float*)d_in[21];
    const float* Wf    = (const float*)d_in[22];
    const float* bf    = (const float*)d_in[23];
    const float* gate  = (const float*)d_in[24];
    const float* Wp    = (const float*)d_in[25];
    const float* bp    = (const float*)d_in[26];
    const float* og    = (const float*)d_in[27];
    const float* ob    = (const float*)d_in[28];
    const float* Wout  = (const float*)d_in[29];
    const float* bout  = (const float*)d_in[30];

    char* wsp = (char*)d_ws;
    auto alloc = [&](size_t bytes) -> void* {
        void* ptr = (void*)wsp;
        wsp += (bytes + 255) & ~(size_t)255;
        return ptr;
    };

    float*  x    = (float*) alloc((size_t)ROWS_X * DIM * 4);
    float*  p    = (float*) alloc((size_t)BB * DIM * 4);
    float*  ha   = (float*) alloc((size_t)ROWS_A * DIM * 4);
    float*  pq   = (float*) alloc((size_t)ROWS_X * DIM * 4);
    float*  pks  = (float*) alloc((size_t)ROWS_X * DIM * 4);
    float*  pvs  = (float*) alloc((size_t)ROWS_X * DIM * 4);
    float*  pka  = (float*) alloc((size_t)ROWS_A * DIM * 4);
    float*  pva  = (float*) alloc((size_t)ROWS_A * DIM * 4);
    float*  pkt  = (float*) alloc((size_t)ROWS_T * DIM * 4);
    float*  pvt  = (float*) alloc((size_t)ROWS_T * DIM * 4);
    __bf16* qh   = (__bf16*)alloc((size_t)ZBH * TQ * HDP * 2);
    __bf16* kh   = (__bf16*)alloc((size_t)ZBH * KEYP * HDP * 2);
    __bf16* vh   = (__bf16*)alloc((size_t)ZBH * KEYP * HDP * 2);
    float*  sc   = (float*) alloc((size_t)ZBH * 64 * KEYP * 4);
    __bf16* at   = (__bf16*)alloc((size_t)ZBH * 64 * KEYP * 2);
    float*  oh   = (float*) alloc((size_t)ZBH * TQ * HDP * 4);
    float*  om   = (float*) alloc((size_t)ROWS_X * DIM * 4);
    float*  o2   = (float*) alloc((size_t)ROWS_X * DIM * 4);
    float*  lnb  = (float*) alloc((size_t)ROWS_X * DIM * 4);

    const long long DD = (long long)DIM * DIM;
    const float scale = 0.09449111825230679f;  // 1/sqrt(112)

    embed_kernel  <<<ROWS_X, 256, 0, stream>>>(toks, temb, x);
    proprio_kernel<<<dim3(4, BB), 256, 0, stream>>>(prop, Wp, bp, p);
    zeropad_kernel<<<(ZBH * 7 * HDP + 255) / 256, 256, 0, stream>>>(kh, vh);

    dim3 g448(DIM / 128, (ROWS_X + 63) / 64);    // 7 x 7
    dim3 g520(DIM / 128, (ROWS_A + 63) / 64);    // 7 x 9
    dim3 g4096(DIM / 128, ROWS_T / 64);          // 7 x 64

    for (int i = 0; i < LAYERS; ++i) {
        const float* mhs_i = mhs + (size_t)i * NTOK * DIM;
        const long long bstride = (long long)LAYERS * NTOK * DIM;

        // projections (bf16 WMMA GEMMs)
        gemm_f32<<<g448, 256, 0, stream>>>(x,  Wq  + i*DD, bq  + i*DIM, pq,  ROWS_X, DIM, DIM, ROWS_X, (long long)ROWS_X*DIM, 0);
        gemm_f32<<<g448, 256, 0, stream>>>(x,  Wks + i*DD, bks + i*DIM, pks, ROWS_X, DIM, DIM, ROWS_X, (long long)ROWS_X*DIM, 0);
        gemm_f32<<<g448, 256, 0, stream>>>(x,  Wvs + i*DD, bvs + i*DIM, pvs, ROWS_X, DIM, DIM, ROWS_X, (long long)ROWS_X*DIM, 0);
        concat_ha_kernel<<<ROWS_A, 256, 0, stream>>>(mhs, p, ha, i);
        gemm_f32<<<g520, 256, 0, stream>>>(ha, Wka + i*DD, bka + i*DIM, pka, ROWS_A, DIM, DIM, ROWS_A, (long long)ROWS_A*DIM, 0);
        gemm_f32<<<g520, 256, 0, stream>>>(ha, Wva + i*DD, bva + i*DIM, pva, ROWS_A, DIM, DIM, ROWS_A, (long long)ROWS_A*DIM, 0);
        gemm_f32<<<g4096, 256, 0, stream>>>(mhs_i, Wkt + i*DD, bkt + i*DIM, pkt, ROWS_T, DIM, DIM, NVIS, bstride, 0);
        gemm_f32<<<g4096, 256, 0, stream>>>(mhs_i, Wvt + i*DD, bvt + i*DIM, pvt, ROWS_T, DIM, DIM, NVIS, bstride, 0);

        // RoPE + head split + bf16 staging (gate folded into task keys)
        stage_rope<<<ROWS_X, 256, 0, stream>>>(pq,  qh, TQ,     TQ,   0,   1, nullptr);
        stage_rope<<<ROWS_X, 256, 0, stream>>>(pks, kh, TQ,     KEYP, 0,   1, nullptr);
        stage_rope<<<ROWS_A, 256, 0, stream>>>(pka, kh, NADP+1, KEYP, 56,  1, nullptr);
        stage_rope<<<ROWS_T, 256, 0, stream>>>(pkt, kh, NVIS,   KEYP, 121, 1, gate + i);
        stage_rope<<<ROWS_X, 256, 0, stream>>>(pvs, vh, TQ,     KEYP, 0,   0, nullptr);
        stage_rope<<<ROWS_A, 256, 0, stream>>>(pva, vh, NADP+1, KEYP, 56,  0, nullptr);
        stage_rope<<<ROWS_T, 256, 0, stream>>>(pvt, vh, NVIS,   KEYP, 121, 0, nullptr);

        // attention: scores = scale * q @ k^T  (batched over B*NH)
        gemm_bf16_batched<<<dim3(KEYP/64, 1, ZBH), 256, 0, stream>>>(
            qh, kh, sc, TQ, KEYP, HDP,
            (long long)TQ * HDP, (long long)KEYP * HDP, (long long)64 * KEYP,
            1, scale);
        softmax_kernel<<<ZBH * TQ, 256, 0, stream>>>(sc, at);
        gemm_bf16_batched<<<dim3(HDP/64, 1, ZBH), 256, 0, stream>>>(
            at, vh, oh, TQ, HDP, KEYP,
            (long long)64 * KEYP, (long long)KEYP * HDP, (long long)TQ * HDP,
            0, 1.0f);

        // output projection, residual LN, feed-forward (relu) -> new x
        merge_heads_kernel<<<ROWS_X, 256, 0, stream>>>(oh, om);
        gemm_f32<<<g448, 256, 0, stream>>>(om,  Wo + i*DD, bo + i*DIM, o2, ROWS_X, DIM, DIM, ROWS_X, (long long)ROWS_X*DIM, 0);
        ln_kernel<<<ROWS_X, 256, 0, stream>>>(o2, x, ln_g + i*DIM, ln_b + i*DIM, lnb);
        gemm_f32<<<g448, 256, 0, stream>>>(lnb, Wf + i*DD, bf + i*DIM, x,  ROWS_X, DIM, DIM, ROWS_X, (long long)ROWS_X*DIM, 1);
    }

    // final LN + output head
    ln_kernel<<<ROWS_X, 256, 0, stream>>>(x, nullptr, og, ob, lnb);
    gemm_f32<<<dim3(VOCAB / 128, (ROWS_X + 63) / 64), 256, 0, stream>>>(
        lnb, Wout, bout, (float*)d_out, ROWS_X, VOCAB, DIM, ROWS_X, (long long)ROWS_X * DIM, 0);
}
